// MultiPauliQCNN_30545807409864
// MI455X (gfx1250) — compile-verified, matrix operand order checked by reading, not run on hardware
//
#include <hip/hip_runtime.h>
#include <math.h>

#define NREG 16  // 512 amplitudes / 32 lanes

typedef float    v2f  __attribute__((ext_vector_type(2)));
typedef float    v8f  __attribute__((ext_vector_type(8)));
typedef _Float16 v16h __attribute__((ext_vector_type(16)));

// ---------------------------------------------------------------------------
// 9-qubit state-vector simulation, one wave32 per patch.
// amp index i = (r << 5) | lane ; qubit q bit = bit q of i.
// qubits 0..4 -> lane bits (cross-lane shuffles), qubits 5..8 -> register bits.
// All conditional updates are branchless (v_cndmask) so no EXEC churn.
// ---------------------------------------------------------------------------

__device__ __forceinline__ int qbit(int q, int lane, int r) {
  return (q < 5) ? ((lane >> q) & 1) : ((r >> (q - 5)) & 1);
}

__device__ __forceinline__ void gate_ry(float re[NREG], float im[NREG], int q,
                                        float th, int lane) {
  const float s = sinf(0.5f * th), c = cosf(0.5f * th);
  if (q >= 5) {
    const int m = 1 << (q - 5);
#pragma unroll
    for (int r = 0; r < NREG; ++r) {
      if (r & m) continue;
      const int r1 = r | m;
      float ar = re[r], ai = im[r], br = re[r1], bi = im[r1];
      re[r]  = c * ar - s * br;  im[r]  = c * ai - s * bi;
      re[r1] = s * ar + c * br;  im[r1] = s * ai + c * bi;
    }
  } else {
    const int bit = (lane >> q) & 1;
    const float sp = bit ? s : -s;  // bit0: c*own - s*partner ; bit1: c*own + s*partner
#pragma unroll
    for (int r = 0; r < NREG; ++r) {
      float pr = __shfl_xor(re[r], 1 << q, 32);
      float pi = __shfl_xor(im[r], 1 << q, 32);
      re[r] = c * re[r] + sp * pr;
      im[r] = c * im[r] + sp * pi;
    }
  }
}

__device__ __forceinline__ void gate_rz(float re[NREG], float im[NREG], int q,
                                        float th, int lane) {
  const float s = sinf(0.5f * th), c = cosf(0.5f * th);
#pragma unroll
  for (int r = 0; r < NREG; ++r) {
    const float sb = qbit(q, lane, r) ? s : -s;  // e^{+-i th/2}
    float nr = c * re[r] - sb * im[r];
    float ni = c * im[r] + sb * re[r];
    re[r] = nr; im[r] = ni;
  }
}

__device__ __forceinline__ void gate_cx(float re[NREG], float im[NREG], int c,
                                        int t, int lane) {
  if (t >= 5) {
    const int tm = 1 << (t - 5);
#pragma unroll
    for (int r = 0; r < NREG; ++r) {
      if (r & tm) continue;
      const int r1 = r | tm;
      const int ctl = qbit(c, lane, r);  // folds to a constant when c >= 5
      float ar = re[r], ai = im[r], br = re[r1], bi = im[r1];
      re[r]  = ctl ? br : ar;  im[r]  = ctl ? bi : ai;
      re[r1] = ctl ? ar : br;  im[r1] = ctl ? ai : bi;
    }
  } else {
#pragma unroll
    for (int r = 0; r < NREG; ++r) {
      float pr = __shfl_xor(re[r], 1 << t, 32);
      float pi = __shfl_xor(im[r], 1 << t, 32);
      const int ctl = qbit(c, lane, r);
      re[r] = ctl ? pr : re[r];
      im[r] = ctl ? pi : im[r];
    }
  }
}

__device__ __forceinline__ void gate_crx(float re[NREG], float im[NREG], int c,
                                         int t, float th, int lane) {
  const float si = sinf(0.5f * th), co = cosf(0.5f * th);
  if (t >= 5) {
    const int tm = 1 << (t - 5);
#pragma unroll
    for (int r = 0; r < NREG; ++r) {
      if (r & tm) continue;
      const int r1 = r | tm;
      const int ctl = qbit(c, lane, r);
      float ar = re[r], ai = im[r], br = re[r1], bi = im[r1];
      // RX: new0 = co*t0 - i si*t1 ; new1 = -i si*t0 + co*t1
      float n0r = co * ar + si * bi, n0i = co * ai - si * br;
      float n1r = co * br + si * ai, n1i = co * bi - si * ar;
      re[r]  = ctl ? n0r : ar;  im[r]  = ctl ? n0i : ai;
      re[r1] = ctl ? n1r : br;  im[r1] = ctl ? n1i : bi;
    }
  } else {
#pragma unroll
    for (int r = 0; r < NREG; ++r) {
      float pr = __shfl_xor(re[r], 1 << t, 32);
      float pi = __shfl_xor(im[r], 1 << t, 32);
      const int ctl = qbit(c, lane, r);
      float nr = co * re[r] + si * pi;   // own + (-i si)*partner (RX symmetric)
      float ni = co * im[r] - si * pr;
      re[r] = ctl ? nr : re[r];
      im[r] = ctl ? ni : im[r];
    }
  }
}

__global__ __launch_bounds__(256) void qcnn_sim_kernel(
    const float* __restrict__ x, const float* __restrict__ w,
    float* __restrict__ feats, int n, int H, int NP) {
  const int lane = threadIdx.x & 31;
  const int wid  = threadIdx.x >> 5;
  const int p    = blockIdx.x * (blockDim.x >> 5) + wid;
  if (p >= NP) return;
  const int b   = p / (n * n);
  const int rem = p - b * n * n;
  const int pi  = rem / n;
  const int pj  = rem - pi * n;

  float ang[9];
#pragma unroll
  for (int di = 0; di < 3; ++di)
#pragma unroll
    for (int dj = 0; dj < 3; ++dj)
      ang[di * 3 + dj] =
          x[(b * H + pi + di) * H + pj + dj] * 3.14159265358979323846f;

  float re[NREG], im[NREG];
#pragma unroll
  for (int r = 0; r < NREG; ++r) { re[r] = 0.0f; im[r] = 0.0f; }
  re[0] = (lane == 0) ? 1.0f : 0.0f;  // |000000000>

  // encoding
#pragma unroll
  for (int q = 0; q < 9; ++q) gate_ry(re, im, q, ang[q], lane);
  // conv1
#pragma unroll
  for (int q = 0; q < 9; ++q) gate_ry(re, im, q, w[q], lane);
#pragma unroll
  for (int q = 0; q < 9; ++q) gate_rz(re, im, q, w[9 + q], lane);
#pragma unroll
  for (int i = 0; i < 9; ++i) gate_cx(re, im, i, (i + 1) % 9, lane);
  // pool1
  gate_crx(re, im, 1, 0, w[18], lane);
  gate_crx(re, im, 2, 0, w[19], lane);
  gate_crx(re, im, 4, 3, w[20], lane);
  gate_crx(re, im, 5, 3, w[21], lane);
  gate_crx(re, im, 7, 6, w[22], lane);
  gate_crx(re, im, 8, 6, w[23], lane);
  // conv2
  gate_ry(re, im, 0, w[24], lane);
  gate_ry(re, im, 3, w[25], lane);
  gate_ry(re, im, 6, w[26], lane);
  gate_rz(re, im, 0, w[27], lane);
  gate_rz(re, im, 3, w[28], lane);
  gate_rz(re, im, 6, w[29], lane);
  gate_cx(re, im, 0, 3, lane);
  gate_cx(re, im, 3, 6, lane);
  gate_cx(re, im, 6, 0, lane);
  gate_ry(re, im, 0, w[30], lane);
  gate_rz(re, im, 3, w[31], lane);
  // pool2
  gate_crx(re, im, 3, 0, w[32], lane);
  gate_crx(re, im, 3, 6, w[33], lane);
  gate_ry(re, im, 0, w[34], lane);
  gate_ry(re, im, 6, w[35], lane);
  // conv3
  gate_ry(re, im, 0, w[36], lane);
  gate_ry(re, im, 6, w[37], lane);
  gate_cx(re, im, 0, 6, lane);
  gate_rz(re, im, 0, w[38], lane);
  gate_rz(re, im, 6, w[39], lane);
  // pool3
  gate_crx(re, im, 6, 0, w[40], lane);
  gate_ry(re, im, 0, w[41], lane);

  // <X>,<Y>,<Z> on qubit 0 (lane bit 0). Each pair counted twice across the
  // two lanes of a pair -> factor 2 in ex/ey cancels.
  const int b0 = lane & 1;
  float tr = 0.0f, ti = 0.0f, ez = 0.0f;
#pragma unroll
  for (int r = 0; r < NREG; ++r) {
    float pr  = __shfl_xor(re[r], 1, 32);
    float pi2 = __shfl_xor(im[r], 1, 32);
    tr += re[r] * pr + im[r] * pi2;             // Re(conj(a0)*a1), symmetric
    float imv = re[r] * pi2 - im[r] * pr;       // Im(conj(own)*partner)
    ti += b0 ? -imv : imv;                      // both halves give Im(conj(a0)*a1)
    float mag = re[r] * re[r] + im[r] * im[r];
    ez += b0 ? -mag : mag;
  }
#pragma unroll
  for (int off = 1; off < 32; off <<= 1) {
    tr += __shfl_xor(tr, off, 32);
    ti += __shfl_xor(ti, off, 32);
    ez += __shfl_xor(ez, off, 32);
  }
  if (lane == 0) {
    feats[3 * p + 0] = tr;
    feats[3 * p + 1] = ti;
    feats[3 * p + 2] = ez;
  }
}

// Zero the shared "zero row" used by out-of-range WMMA lanes.
__global__ void zero_kernel(float* __restrict__ z, int count) {
  const int i = blockIdx.x * blockDim.x + threadIdx.x;
  if (i < count) z[i] = 0.0f;
}

// ---------------------------------------------------------------------------
// FC: out(8,10) = feats(8,2028) @ fc_w(10,2028)^T + fc_b, via WMMA (one wave).
// Row guards hoisted to a per-lane row pointer (out-of-range -> zero row), so
// the K-loop is branchless: EXEC stays all-1s (WMMA requirement) and loads
// merge into contiguous b64 accesses. K % 4 == 0, so no K guard.
// ---------------------------------------------------------------------------
__global__ void fc_kernel(const float* __restrict__ feats,
                          const float* __restrict__ fc_w,
                          const float* __restrict__ fc_b,
                          const float* __restrict__ zrow,
                          float* __restrict__ out, int Mrows, int K, int N) {
#if __has_builtin(__builtin_amdgcn_wmma_f32_16x16x4_f32)
  const int lane = threadIdx.x & 31;
  const int row = lane & 15;            // M row for A, N col for B
  const int kh  = (lane >> 4) << 1;     // K half-select: 0 or 2
  const float* arow = (row < Mrows) ? (feats + row * K) : zrow;
  const float* brow = (row < N)     ? (fc_w + row * K)  : zrow;
  v8f acc = {};
#pragma unroll 4
  for (int k0 = 0; k0 < K; k0 += 4) {
    v2f a, bm;
    a[0]  = arow[k0 + kh];  a[1]  = arow[k0 + kh + 1];
    bm[0] = brow[k0 + kh];  bm[1] = brow[k0 + kh + 1];
    acc = __builtin_amdgcn_wmma_f32_16x16x4_f32(false, a, false, bm, (short)0,
                                                acc, false, false);
  }
  const int nn = lane & 15;
  const int mb = (lane >> 4) << 3;
#pragma unroll
  for (int r = 0; r < 8; ++r) {
    const int m = mb + r;
    if (m < Mrows && nn < N) out[m * N + nn] = acc[r] + fc_b[nn];
  }
#elif __has_builtin(__builtin_amdgcn_wmma_f32_16x16x32_f16)
  const int lane = threadIdx.x & 31;
  const int row = lane & 15;
  const int h   = lane >> 4;
  const float* arow = (row < Mrows) ? (feats + row * K) : zrow;
  const float* brow = (row < N)     ? (fc_w + row * K)  : zrow;
  v8f acc = {};
  for (int k0 = 0; k0 < K; k0 += 32) {
    v16h a = {}, bm = {};
#pragma unroll
    for (int e = 0; e < 16; ++e) {
      const int v = e >> 1, odd = e & 1;
      const int kk = k0 + ((v & 4) ? 16 : 0) + h * 8 + (v & 3) * 2 + odd;
      const bool kin = kk < K;
      a[e]  = kin ? (_Float16)arow[kk] : (_Float16)0.0f;
      bm[e] = kin ? (_Float16)brow[kk] : (_Float16)0.0f;
    }
    acc = __builtin_amdgcn_wmma_f32_16x16x32_f16(false, a, false, bm, (short)0,
                                                 acc, false, false);
  }
  const int nn = lane & 15;
  const int mb = (lane >> 4) << 3;
#pragma unroll
  for (int r = 0; r < 8; ++r) {
    const int m = mb + r;
    if (m < Mrows && nn < N) out[m * N + nn] = acc[r] + fc_b[nn];
  }
#else
  // Host-pass / safety fallback.
  const int idx = threadIdx.x;
  if (idx < N) {
    for (int m = 0; m < Mrows; ++m) {
      float s = fc_b[idx];
      for (int k = 0; k < K; ++k) s += feats[m * K + k] * fc_w[idx * K + k];
      out[m * N + idx] = s;
    }
  }
#endif
}

extern "C" void kernel_launch(void* const* d_in, const int* in_sizes, int n_in,
                              void* d_out, int out_size, void* d_ws,
                              size_t ws_size, hipStream_t stream) {
  const float* x    = (const float*)d_in[0];  // (B,1,28,28)
  const float* w    = (const float*)d_in[1];  // (42,)
  const float* fc_w = (const float*)d_in[2];  // (10, 2028)
  const float* fc_b = (const float*)d_in[3];  // (10,)
  float* out = (float*)d_out;                 // (B, 10)

  const int H  = 28;
  const int B  = in_sizes[0] / (H * H);
  const int n  = H - 3 + 1;                   // 26
  const int NP = B * n * n;                   // 5408 patches
  const int K  = n * n * 3;                   // 2028
  const int N  = out_size / B;                // 10

  float* feats = (float*)d_ws;                // NP*3 floats
  float* zrow  = feats + NP * 3;              // K zero floats (8-byte aligned: NP*3 even)

  zero_kernel<<<(K + 255) / 256, 256, 0, stream>>>(zrow, K);

  const int wavesPerBlock = 8;                // 256 threads = 8 wave32
  dim3 block(32 * wavesPerBlock);
  dim3 grid((NP + wavesPerBlock - 1) / wavesPerBlock);
  qcnn_sim_kernel<<<grid, block, 0, stream>>>(x, w, feats, n, H, NP);

  fc_kernel<<<1, 32, 0, stream>>>(feats, fc_w, fc_b, zrow, out, B, K, N);
}
